// ResidualQuantizer_65266323030552
// MI455X (gfx1250) — compile-verified, matrix operand order checked by reading, not run on hardware
//
#include <hip/hip_runtime.h>
#include <stdint.h>

typedef __attribute__((ext_vector_type(16))) __bf16 v16bf;
typedef __attribute__((ext_vector_type(8)))  float  v8f;

#define RQ_N 32768
#define RQ_D 256
#define RQ_K 8192
#define RQ_L 3
#define RQ_CK 8          // D/32 k-chunks
#define RQ_TILES 512     // K/16 centroid tiles

// Mapping of the 16 packed bf16 elements in one lane of a 16-bit 16x32 WMMA
// fragment to the k index (ISA 7.12.2: lanes 0-15 hold K=0..7 in V0-3 and
// K=16..23 in V4-7; lanes 16-31 hold K=8..15 / K=24..31).
__device__ __forceinline__ int rq_klocal(int lane, int j) {
  int half = (lane >> 4) & 1;
  int v = j >> 1;
  int kk = (v < 4) ? (2 * v + (j & 1)) : (16 + 2 * (v - 4) + (j & 1));
  return kk + half * 8;
}

// ---------------------------------------------------------------------------
// Split a row-major (nrows x 256) fp32 matrix into bf16 hi/lo planes laid out
// in WMMA-fragment order: elem = ((rb*8 + ck)*32 + lane)*16 + j,
// source = src[(rb*16 + (lane&15))*256 + ck*32 + klocal(lane,j)].
// Used for both the residual (A fragments) and the codebook (B fragments).
// ---------------------------------------------------------------------------
__global__ void __launch_bounds__(256) rq_split_kernel(
    const float* __restrict__ src, __bf16* __restrict__ Hh,
    __bf16* __restrict__ Hl) {
  int gid  = blockIdx.x * 256 + threadIdx.x;
  int j    = gid & 15;
  int lane = (gid >> 4) & 31;
  int ck   = (gid >> 9) & 7;
  int rb   = gid >> 12;
  int row  = rb * 16 + (lane & 15);
  int d    = ck * 32 + rq_klocal(lane, j);
  float x  = src[(size_t)row * RQ_D + d];
  __bf16 h = (__bf16)x;
  __bf16 l = (__bf16)(x - (float)h);
  Hh[gid] = h;
  Hl[gid] = l;
}

// ||c||^2 per centroid (fp32, exact path for the score bias).
__global__ void __launch_bounds__(256) rq_cnorm_kernel(
    const float* __restrict__ C, float* __restrict__ cnorm) {
  int k = blockIdx.x * 256 + threadIdx.x;
  const float* row = C + (size_t)k * RQ_D;
  float s = 0.f;
  for (int d = 0; d < RQ_D; ++d) { float v = row[d]; s += v * v; }
  cnorm[k] = s;
}

// ---------------------------------------------------------------------------
// Async-stage one 16-centroid B tile (hi+lo bf16 fragments, 16KB) into LDS.
// 128 threads move 128B each with global_load_async_to_lds_b128
// (INST_OFFSET advances both the global and the LDS address).
// ---------------------------------------------------------------------------
__device__ __forceinline__ void rq_stage_tile(const __bf16* CBh,
                                              const __bf16* CBl, int tile,
                                              int buf, unsigned tid,
                                              unsigned short* smem) {
  unsigned ldsbase = (unsigned)(uintptr_t)(&smem[0]);
  unsigned ldsh = ldsbase + (unsigned)buf * 16384u + tid * 64u;
  unsigned ldsl = ldsh + 8192u;
  unsigned voff = tid * 64u;
  uint64_t gh = (uint64_t)(uintptr_t)CBh + (uint64_t)tile * 8192u;
  uint64_t gl = (uint64_t)(uintptr_t)CBl + (uint64_t)tile * 8192u;
  asm volatile(
      "global_load_async_to_lds_b128 %0, %1, %2\n\t"
      "global_load_async_to_lds_b128 %0, %1, %2 offset:16\n\t"
      "global_load_async_to_lds_b128 %0, %1, %2 offset:32\n\t"
      "global_load_async_to_lds_b128 %0, %1, %2 offset:48\n\t"
      "global_load_async_to_lds_b128 %3, %1, %4\n\t"
      "global_load_async_to_lds_b128 %3, %1, %4 offset:16\n\t"
      "global_load_async_to_lds_b128 %3, %1, %4 offset:32\n\t"
      "global_load_async_to_lds_b128 %3, %1, %4 offset:48"
      :
      : "v"(ldsh), "v"(voff), "s"(gh), "v"(ldsl), "s"(gl)
      : "memory");
}

// ---------------------------------------------------------------------------
// Fused GEMM (scores = ||c||^2 - 2 r.c) + row argmin over all 8192 centroids.
// Grid: N/128 blocks, 128 threads (4 waves). Wave w owns 32 rows (two 16-row
// A fragment sets kept in registers: ~256 VGPRs) so each 16KB LDS B-tile read
// feeds 48 bf16 WMMAs — balancing LDS bandwidth against the XDL rate.
// B tiles are double-buffered in LDS via async global->LDS loads.
// ---------------------------------------------------------------------------
__global__ void __launch_bounds__(128) rq_gemm_argmin_kernel(
    const __bf16* __restrict__ Rh, const __bf16* __restrict__ Rl,
    const __bf16* __restrict__ CBh, const __bf16* __restrict__ CBl,
    const float* __restrict__ cnorm, int* __restrict__ idx_out) {
  __shared__ __align__(32) unsigned short smem[16384];  // 2 bufs x 16KB

  const unsigned tid = threadIdx.x;
  const int lane = tid & 31;
  const int wave = tid >> 5;                  // 0..3
  const int rb0 = blockIdx.x * 8 + wave * 2;  // first 16-row fragment block
  const int col = lane & 15;

  // A fragments in registers: two 16-row blocks x 8 chunks x (hi, lo).
  v16bf Ah0[RQ_CK], Al0[RQ_CK], Ah1[RQ_CK], Al1[RQ_CK];
#pragma unroll
  for (int ck = 0; ck < RQ_CK; ++ck) {
    size_t o0 = ((size_t)(rb0 * RQ_CK + ck) * 32 + lane) * 16;
    size_t o1 = ((size_t)((rb0 + 1) * RQ_CK + ck) * 32 + lane) * 16;
    Ah0[ck] = *(const v16bf*)(Rh + o0);
    Al0[ck] = *(const v16bf*)(Rl + o0);
    Ah1[ck] = *(const v16bf*)(Rh + o1);
    Al1[ck] = *(const v16bf*)(Rl + o1);
  }

  float bs0[8], bs1[8];
  int bi0[8], bi1[8];
#pragma unroll
  for (int g = 0; g < 8; ++g) {
    bs0[g] = 3.4e38f; bi0[g] = 0x7fffffff;
    bs1[g] = 3.4e38f; bi1[g] = 0x7fffffff;
  }

  rq_stage_tile(CBh, CBl, 0, 0, tid, smem);
  asm volatile("s_wait_asynccnt 0" ::: "memory");
  __syncthreads();

  for (int t = 0; t < RQ_TILES; ++t) {
    const int buf = t & 1;
    if (t + 1 < RQ_TILES) rq_stage_tile(CBh, CBl, t + 1, buf ^ 1, tid, smem);

    float cn = cnorm[t * 16 + col];
    v8f acc0 = {0.f, 0.f, 0.f, 0.f, 0.f, 0.f, 0.f, 0.f};
    v8f acc1 = {0.f, 0.f, 0.f, 0.f, 0.f, 0.f, 0.f, 0.f};
#pragma unroll
    for (int ck = 0; ck < RQ_CK; ++ck) {
      const v16bf bh =
          *(const v16bf*)&smem[buf * 8192 + 0 * 4096 + ck * 512 + lane * 16];
      const v16bf bl =
          *(const v16bf*)&smem[buf * 8192 + 1 * 4096 + ck * 512 + lane * 16];
      acc0 = __builtin_amdgcn_wmma_f32_16x16x32_bf16(false, Ah0[ck], false, bh,
                                                     (short)0, acc0, false, false);
      acc1 = __builtin_amdgcn_wmma_f32_16x16x32_bf16(false, Ah1[ck], false, bh,
                                                     (short)0, acc1, false, false);
      acc0 = __builtin_amdgcn_wmma_f32_16x16x32_bf16(false, Al0[ck], false, bh,
                                                     (short)0, acc0, false, false);
      acc1 = __builtin_amdgcn_wmma_f32_16x16x32_bf16(false, Al1[ck], false, bh,
                                                     (short)0, acc1, false, false);
      acc0 = __builtin_amdgcn_wmma_f32_16x16x32_bf16(false, Ah0[ck], false, bl,
                                                     (short)0, acc0, false, false);
      acc1 = __builtin_amdgcn_wmma_f32_16x16x32_bf16(false, Ah1[ck], false, bl,
                                                     (short)0, acc1, false, false);
    }

    const int n = t * 16 + col;
#pragma unroll
    for (int g = 0; g < 8; ++g) {
      float s0 = cn - 2.0f * acc0[g];
      float s1 = cn - 2.0f * acc1[g];
      if (s0 < bs0[g] || (s0 == bs0[g] && n < bi0[g])) { bs0[g] = s0; bi0[g] = n; }
      if (s1 < bs1[g] || (s1 == bs1[g] && n < bi1[g])) { bs1[g] = s1; bi1[g] = n; }
    }

    asm volatile("s_wait_asynccnt 0" ::: "memory");
    __syncthreads();
  }

  // Reduce (min, argmin) across the 16 lanes of each wave half; slot g in
  // lanes 0-15 holds row g, in lanes 16-31 row 8+g (C/D layout).
#pragma unroll
  for (int g = 0; g < 8; ++g) {
    for (int m = 1; m < 16; m <<= 1) {
      float os0 = __shfl_xor(bs0[g], m, 32);
      int oi0 = __shfl_xor(bi0[g], m, 32);
      if (os0 < bs0[g] || (os0 == bs0[g] && oi0 < bi0[g])) {
        bs0[g] = os0; bi0[g] = oi0;
      }
      float os1 = __shfl_xor(bs1[g], m, 32);
      int oi1 = __shfl_xor(bi1[g], m, 32);
      if (os1 < bs1[g] || (os1 == bs1[g] && oi1 < bi1[g])) {
        bs1[g] = os1; bi1[g] = oi1;
      }
    }
    if (col == 0) {
      int row0 = rb0 * 16 + (lane >> 4) * 8 + g;
      idx_out[row0] = bi0[g];
      idx_out[row0 + 16] = bi1[g];
    }
  }
}

// ---------------------------------------------------------------------------
// Residual update: r_new = prev - C[idx].  Driven in fragment coordinates so
// it can also emit the next level's bf16 hi/lo A planes.  level==2 instead
// writes e_st = e - r_new (== e_q).  Also writes codes (exact small ints as
// float) once per row.
// ---------------------------------------------------------------------------
__global__ void __launch_bounds__(256) rq_update_kernel(
    const float* __restrict__ prev, const float* __restrict__ C,
    const int* __restrict__ idx, int level, float* __restrict__ r_out,
    __bf16* __restrict__ Rh, __bf16* __restrict__ Rl,
    float* __restrict__ codes_out, const float* __restrict__ e,
    float* __restrict__ est_out) {
  int gid  = blockIdx.x * 256 + threadIdx.x;
  int j    = gid & 15;
  int lane = (gid >> 4) & 31;
  int ck   = (gid >> 9) & 7;
  int rb   = gid >> 12;
  int row  = rb * 16 + (lane & 15);
  int d    = ck * 32 + rq_klocal(lane, j);

  int k = idx[row];
  size_t rd = (size_t)row * RQ_D + d;
  float val = prev[rd] - C[(size_t)k * RQ_D + d];

  if (level < 2) {
    r_out[rd] = val;
    __bf16 h = (__bf16)val;
    Rh[gid] = h;
    Rl[gid] = (__bf16)(val - (float)h);
  } else {
    est_out[rd] = e[rd] - val;  // e - r_final == e_q == e_st
  }
  if (ck == 0 && lane < 16 && j == 0) {
    codes_out[(size_t)row * RQ_L + level] = (float)k;
  }
}

// ---------------------------------------------------------------------------
extern "C" void kernel_launch(void* const* d_in, const int* in_sizes, int n_in,
                              void* d_out, int out_size, void* d_ws,
                              size_t ws_size, hipStream_t stream) {
  const float* e  = (const float*)d_in[0];  // (32768, 256)
  const float* cb = (const float*)d_in[1];  // (3, 8192, 256)

  float* out_codes = (float*)d_out;                        // N*3 (ints as f32)
  float* out_est   = (float*)d_out + (size_t)RQ_N * RQ_L;  // N*D

  char* ws = (char*)d_ws;
  size_t off = 0;
  float*  r   = (float*)(ws + off);  off += (size_t)RQ_N * RQ_D * 4;  // 32MB
  __bf16* Rh  = (__bf16*)(ws + off); off += (size_t)RQ_N * RQ_D * 2;  // 16MB
  __bf16* Rl  = (__bf16*)(ws + off); off += (size_t)RQ_N * RQ_D * 2;  // 16MB
  __bf16* CBh = (__bf16*)(ws + off); off += (size_t)RQ_K * RQ_D * 2;  // 4MB
  __bf16* CBl = (__bf16*)(ws + off); off += (size_t)RQ_K * RQ_D * 2;  // 4MB
  float*  cn  = (float*)(ws + off);  off += (size_t)RQ_K * 4;         // 32KB
  int*    idx = (int*)(ws + off);    off += (size_t)RQ_N * 4;         // 128KB

  // Level-0 residual is e itself: build its bf16 hi/lo A planes.
  rq_split_kernel<<<(RQ_N * RQ_D) / 256, 256, 0, stream>>>(e, Rh, Rl);

  for (int l = 0; l < RQ_L; ++l) {
    const float* Cl = cb + (size_t)l * RQ_K * RQ_D;
    rq_cnorm_kernel<<<RQ_K / 256, 256, 0, stream>>>(Cl, cn);
    rq_split_kernel<<<(RQ_K * RQ_D) / 256, 256, 0, stream>>>(Cl, CBh, CBl);
    rq_gemm_argmin_kernel<<<RQ_N / 128, 128, 0, stream>>>(Rh, Rl, CBh, CBl,
                                                          cn, idx);
    rq_update_kernel<<<(RQ_N * RQ_D) / 256, 256, 0, stream>>>(
        (l == 0) ? e : r, Cl, idx, l, r, Rh, Rl, out_codes, e, out_est);
  }
}
